// FinancialRNN_37005438222678
// MI455X (gfx1250) — compile-verified, matrix operand order checked by reading, not run on hardware
//
#include <hip/hip_runtime.h>

typedef _Float16 v16h __attribute__((ext_vector_type(16)));
typedef _Float16 v8h  __attribute__((ext_vector_type(8)));
typedef float    v8f  __attribute__((ext_vector_type(8)));

namespace {
constexpr int kF  = 64;        // input features
constexpr int kH  = 32;        // hidden size
constexpr int kG  = 4 * kH;    // 128 gate columns (i,f,g,o)
constexpr int kBT = 16;        // batch rows per workgroup (WMMA M)
}

union Frag {
  v16h v;
  v8h  h[2];
};

// Branch-free fast activations: v_exp_f32 + v_rcp_f32, no div_fixup path,
// no EXEC divergence (keeps the latency-bound scan's critical path short).
__device__ __forceinline__ float fsigmoid(float v) {
  return __builtin_amdgcn_rcpf(1.0f + __expf(-v));
}
__device__ __forceinline__ float ftanh(float v) {
  // tanh(x) = 1 - 2/(exp(2x)+1); saturates to +/-1 for large |x|.
  return __builtin_fmaf(-2.0f, __builtin_amdgcn_rcpf(__expf(2.0f * v) + 1.0f),
                        1.0f);
}

// Load the two 16x32 (K-split of 16x64) f16 A-fragments of x[row, t, :]
// directly from global, per ISA 16-bit A layout:
//   lanes 0-15  : K = kbase..kbase+7  and kbase+16..kbase+23  (kbase = 0)
//   lanes 16-31 : same with kbase = 8
__device__ __forceinline__ void load_x_frags(const float* __restrict__ x,
                                             int row, int T, int t, int kbase,
                                             Frag& a0, Frag& a1) {
  const float* p = x + ((size_t)row * T + t) * kF;
#pragma unroll
  for (int j = 0; j < 8; ++j) {
    a0.v[j]     = (_Float16)p[kbase + j];
    a0.v[j + 8] = (_Float16)p[kbase + 16 + j];
    a1.v[j]     = (_Float16)p[kbase + 32 + j];
    a1.v[j + 8] = (_Float16)p[kbase + 48 + j];
  }
}

__global__ __launch_bounds__(256, 1) void lstm_fused_kernel(
    const float* __restrict__ x,   // [B, T, 64]
    const float* __restrict__ Wx,  // [64, 128]
    const float* __restrict__ Wh,  // [32, 128]
    const float* __restrict__ b,   // [128]
    float* __restrict__ out,       // [B, T, 32]
    int T) {
  __shared__ __align__(16) _Float16 h16[kBT][kH];   // 1 KB: h_t as next A-frag
  __shared__ __align__(16) float    gates[kBT][kG]; // 8 KB

  const int tid   = threadIdx.x;
  const int wave  = tid >> 5;           // 0..7 -> gate-column tile
  const int lane  = tid & 31;
  const int lhalf = lane >> 4;          // 0 | 1
  const int lmod  = lane & 15;          // row M (A) / col N (B,C)
  const int kbase = lhalf * 8;
  const int row0  = blockIdx.x * kBT;
  const int ncol  = wave * 16 + lmod;   // gate column 0..127

  // ---- one-time: B-matrix fragments (f16) in registers, reused for all T ----
  // B layout (mirror of A): lanes 0-15 hold K in {0..7,16..23}, lanes 16-31
  // hold K in {8..15,24..31}, column N = lmod within this wave's 16-col tile.
  v16h whB, wxB0, wxB1;
#pragma unroll
  for (int j = 0; j < 8; ++j) {
    const int k0 = kbase + j;
    const int k1 = kbase + 16 + j;
    whB[j]      = (_Float16)Wh[k0 * kG + ncol];
    whB[j + 8]  = (_Float16)Wh[k1 * kG + ncol];
    wxB0[j]     = (_Float16)Wx[k0 * kG + ncol];
    wxB0[j + 8] = (_Float16)Wx[k1 * kG + ncol];
    wxB1[j]     = (_Float16)Wx[(k0 + 32) * kG + ncol];
    wxB1[j + 8] = (_Float16)Wx[(k1 + 32) * kG + ncol];
  }
  const float bias = b[ncol];

  // ---- pointwise ownership: each thread owns 2 (m, n) cells; c in registers
  const int pn  = tid & 31;        // hidden column
  const int pm0 = tid >> 5;        // rows 0..7
  const int pm1 = 8 + pm0;         // rows 8..15
  float cst0 = 0.0f, cst1 = 0.0f;

  // h_0 = 0
  ((_Float16*)h16)[tid]       = (_Float16)0.0f;
  ((_Float16*)h16)[tid + 256] = (_Float16)0.0f;

  // software pipeline: x fragments for step t preloaded in registers
  Frag xA0, xA1, nxA0, nxA1;
  load_x_frags(x, row0 + lmod, T, 0, kbase, xA0, xA1);

  for (int t = 0; t < T; ++t) {
    __syncthreads();  // h16(t-1) writes + gates(t-1) reads complete

    // A-fragment of h_{t-1} (16x32 f16) from LDS: two 16B chunks per lane
    Frag hA;
    hA.h[0] = *(const v8h*)&h16[lmod][kbase];
    hA.h[1] = *(const v8h*)&h16[lmod][kbase + 16];

    // gates tile = b + h@Wh + x@Wx  (f32 accumulate, 3 chained WMMAs)
    v8f acc;
#pragma unroll
    for (int r = 0; r < 8; ++r) acc[r] = bias;
    acc = __builtin_amdgcn_wmma_f32_16x16x32_f16(false, hA.v,  false, whB,
                                                 (short)0, acc, false, false);
    acc = __builtin_amdgcn_wmma_f32_16x16x32_f16(false, xA0.v, false, wxB0,
                                                 (short)0, acc, false, false);
    acc = __builtin_amdgcn_wmma_f32_16x16x32_f16(false, xA1.v, false, wxB1,
                                                 (short)0, acc, false, false);

    // C layout: VGPR r holds M = r (lanes 0-15) / r+8 (lanes 16-31), N = lmod
#pragma unroll
    for (int r = 0; r < 8; ++r) gates[r + 8 * lhalf][ncol] = acc[r];

    // overlap: fetch next step's x fragments while gates settle
    if (t + 1 < T) load_x_frags(x, row0 + lmod, T, t + 1, kbase, nxA0, nxA1);

    __syncthreads();  // gates(t) visible

    // pointwise LSTM cell update (flax gate order i,f,g,o), branch-free
    {
      const float gi = gates[pm0][pn];
      const float gf = gates[pm0][pn + kH];
      const float gg = gates[pm0][pn + 2 * kH];
      const float go = gates[pm0][pn + 3 * kH];
      const float cn = fsigmoid(gf) * cst0 + fsigmoid(gi) * ftanh(gg);
      const float hn = fsigmoid(go) * ftanh(cn);
      cst0 = cn;
      h16[pm0][pn] = (_Float16)hn;
      out[((size_t)(row0 + pm0) * T + t) * kH + pn] = hn;
    }
    {
      const float gi = gates[pm1][pn];
      const float gf = gates[pm1][pn + kH];
      const float gg = gates[pm1][pn + 2 * kH];
      const float go = gates[pm1][pn + 3 * kH];
      const float cn = fsigmoid(gf) * cst1 + fsigmoid(gi) * ftanh(gg);
      const float hn = fsigmoid(go) * ftanh(cn);
      cst1 = cn;
      h16[pm1][pn] = (_Float16)hn;
      out[((size_t)(row0 + pm1) * T + t) * kH + pn] = hn;
    }

    xA0 = nxA0;
    xA1 = nxA1;
  }
}

extern "C" void kernel_launch(void* const* d_in, const int* in_sizes, int n_in,
                              void* d_out, int out_size, void* d_ws,
                              size_t ws_size, hipStream_t stream) {
  const float* x  = (const float*)d_in[0];  // [B, T, 64]
  const float* Wx = (const float*)d_in[1];  // [64, 128]
  const float* Wh = (const float*)d_in[2];  // [32, 128]
  const float* b  = (const float*)d_in[3];  // [128]
  float* out = (float*)d_out;               // [B, T, 32]

  const int T = 2048;
  const int B = in_sizes[0] / (T * kF);     // 256
  dim3 grid(B / kBT);                       // 16 workgroups, 16 batch rows each
  dim3 block(256);                          // 8 wave32s = 8 gate-column tiles
  lstm_fused_kernel<<<grid, block, 0, stream>>>(x, Wx, Wh, b, out, T);
}
